// MixedAttention_3865470566919
// MI455X (gfx1250) — matrix operand order repaired and checked by reading, not compiled, not run on hardware
//
#include <hip/hip_runtime.h>
#include <hip/hip_bf16.h>

typedef unsigned short u16;
typedef unsigned int   u32;

typedef __attribute__((ext_vector_type(16))) __bf16 v16bf;
typedef __attribute__((ext_vector_type(8)))  float  v8f;

// ---------- bf16 helpers (bit-level, no __bf16 arithmetic needed) ----------
__device__ __forceinline__ u16 f2bf(float f) {
  u32 u = __float_as_uint(f);
  u32 r = u + 0x7FFFu + ((u >> 16) & 1u);   // round-to-nearest-even
  return (u16)(r >> 16);
}
__device__ __forceinline__ float bf2f(u16 s) { return __uint_as_float(((u32)s) << 16); }
__device__ __forceinline__ float bflo(u32 u) { return __uint_as_float(u << 16); }
__device__ __forceinline__ float bfhi(u32 u) { return __uint_as_float(u & 0xFFFF0000u); }

// ------------------------------ constants ---------------------------------
#define LTOK 65536
#define DIM 192
#define HEADS 6
#define LOG100 4.6051702f

// ------------------------- f32 -> bf16 convert ----------------------------
__global__ __launch_bounds__(256) void k_convert(const float* __restrict__ src,
                                                 u16* __restrict__ dst, int n) {
  int i = blockIdx.x * blockDim.x + threadIdx.x;
  if (i < n) dst[i] = f2bf(src[i]);
}

// ---------- f32 [K x N] -> bf16 transposed+padded [Npad x K] --------------
__global__ __launch_bounds__(256) void k_convert_T(const float* __restrict__ src,
                                                   u16* __restrict__ dst,
                                                   int Kdim, int N, int Npad) {
  int i = blockIdx.x * blockDim.x + threadIdx.x;
  if (i >= Npad * Kdim) return;
  int n = i / Kdim, k = i % Kdim;
  float v = (n < N) ? src[(size_t)k * N + n] : 0.0f;
  dst[i] = f2bf(v);
}

// --------------------------- WMMA bf16 GEMM -------------------------------
// C[M,N] = A[M,K](bf16,row) * Bt[Npad,K](bf16, N-major) + bias[N]
// Block: 256 thr = 8 waves; tile M=128, N=64, K-step 32; wave -> 32x32 via 2x2 WMMA.
// Double-buffered LDS staging; K templated so the chunk loop fully unrolls.
#define BM 128
#define BN 64
#define BK 32
#define APAD 40   // LDS row stride (ushorts): 80B rows, 16B aligned, conflict-friendly

union FragBF { v16bf v; u32 u[8]; };
union AccF   { v8f  v; float f[8]; };

template <int K>
__global__ __launch_bounds__(256) void k_gemm_bf16(
    const u16* __restrict__ A, int lda,
    const u16* __restrict__ Bt,          // Npad x K, N-major, zero padded
    const float* __restrict__ bias,
    float* __restrict__ Cf, u16* __restrict__ Cb,
    int ldc, int M, int N) {
  __shared__ u16 As[2][BM * APAD];
  __shared__ u16 Bs[2][BN * APAD];

  int tid  = threadIdx.x;
  int lane = tid & 31;
  int wv   = tid >> 5;
  int wm   = wv & 3;           // 4 waves along M
  int wn   = wv >> 2;          // 2 waves along N
  int bm0  = blockIdx.x * BM;
  int bn0  = blockIdx.y * BN;
  int half = lane >> 4;
  int l15  = lane & 15;

  // Per-thread staging coordinates (all vectorized b128 transfers).
  int ar  = tid >> 1;                 // A: row in tile
  int acb = (tid & 1) * 16;           // A: 16 contiguous bf16
  const u16* a_src0 = A + (size_t)(bm0 + ar) * lda + acb;
  int bn  = tid >> 2;                 // B: n-row in tile
  int bcb = (tid & 3) * 8;            // B: 8 contiguous bf16 (one uint4)
  const u16* b_src0 = Bt + (size_t)(bn0 + bn) * K + bcb;

  auto stage = [&](int b, int k0) {
    const uint4* sa = (const uint4*)(a_src0 + k0);
    uint4* da = (uint4*)(&As[b][ar * APAD + acb]);
    da[0] = sa[0];
    da[1] = sa[1];
    *(uint4*)(&Bs[b][bn * APAD + bcb]) = *(const uint4*)(b_src0 + k0);
  };

  AccF acc[2][2];
  #pragma unroll
  for (int i = 0; i < 2; i++)
    #pragma unroll
    for (int j = 0; j < 2; j++)
      #pragma unroll
      for (int r = 0; r < 8; r++) acc[i][j].f[r] = 0.0f;

  constexpr int NK = K / BK;
  stage(0, 0);

  #pragma unroll
  for (int kc = 0; kc < NK; kc++) {
    __syncthreads();
    if (kc + 1 < NK) stage((kc + 1) & 1, (kc + 1) * BK);
    int b = kc & 1;

    // Fragments per ISA 16-bit A(16x32) / B(32x16) layouts.
    FragBF af[2], bf[2];
    #pragma unroll
    for (int mt = 0; mt < 2; mt++) {
      int row = wm * 32 + mt * 16 + l15;
      #pragma unroll
      for (int p = 0; p < 8; p++) {
        int kb = ((p < 4) ? (2 * p) : (16 + 2 * (p - 4))) + half * 8;
        af[mt].u[p] = *(const u32*)(&As[b][row * APAD + kb]);
      }
    }
    #pragma unroll
    for (int nt = 0; nt < 2; nt++) {
      int col = wn * 32 + nt * 16 + l15;
      int kbB = half * 8;
      #pragma unroll
      for (int p = 0; p < 8; p++)
        bf[nt].u[p] = *(const u32*)(&Bs[b][col * APAD + kbB + ((p < 4) ? (2 * p) : (16 + 2 * (p - 4)))]);
    }
    #pragma unroll
    for (int mt = 0; mt < 2; mt++)
      #pragma unroll
      for (int nt = 0; nt < 2; nt++)
        acc[mt][nt].v = __builtin_amdgcn_wmma_f32_16x16x32_bf16(
            false, af[mt].v, false, bf[nt].v, (short)0, acc[mt][nt].v, false, false);
  }

  // Epilogue: C/D layout (lane<16: M=r, N=lane; lane>=16: M=r+8).
  #pragma unroll
  for (int mt = 0; mt < 2; mt++) {
    #pragma unroll
    for (int nt = 0; nt < 2; nt++) {
      int col = bn0 + wn * 32 + nt * 16 + l15;
      if (col >= N) continue;
      float bv = bias ? bias[col] : 0.0f;
      #pragma unroll
      for (int r = 0; r < 8; r++) {
        int row = bm0 + wm * 32 + mt * 16 + (half ? (r + 8) : r);
        float v = acc[mt][nt].f[r] + bv;
        size_t o = (size_t)row * ldc + col;
        if (Cf) Cf[o] = v;
        if (Cb) Cb[o] = f2bf(v);
      }
    }
  }
}

// ------------------- anchor: depthwise 3x3 s2 + pointwise -----------------
__global__ __launch_bounds__(192) void k_anchor(const float* __restrict__ x,
                                                const float* __restrict__ dww,
                                                const float* __restrict__ dwb,
                                                const float* __restrict__ pww,
                                                const float* __restrict__ pwb,
                                                float* __restrict__ outa) {
  __shared__ float dw[DIM];
  int pix = blockIdx.x;
  int oh = pix >> 7, ow = pix & 127;
  int c = threadIdx.x;
  float acc = dwb[c];
  #pragma unroll
  for (int ky = 0; ky < 3; ky++) {
    int ih = 2 * oh - 1 + ky;
    if (ih < 0 || ih > 255) continue;
    #pragma unroll
    for (int kx = 0; kx < 3; kx++) {
      int iw = 2 * ow - 1 + kx;
      if (iw < 0 || iw > 255) continue;
      acc += x[((size_t)(ih * 256 + iw)) * DIM + c] * dww[(ky * 3 + kx) * DIM + c];
    }
  }
  dw[c] = acc;
  __syncthreads();
  if (c < 96) {
    float o = pwb[c];
    #pragma unroll 4
    for (int ci = 0; ci < DIM; ci++) o += dw[ci] * pww[ci * 96 + c];
    outa[(size_t)pix * 96 + c] = o;
  }
}

// ----------------- neighborhood attention (K=7, dil=2) --------------------
__device__ __forceinline__ void nbr7(int i, int* nb, int* rb) {
  int g = i & 1, p = i >> 1;
  int start = p - 3;
  if (start < 0) start = 0;
  if (start > 121) start = 121;   // gsize - K = 128 - 7
  #pragma unroll
  for (int j = 0; j < 7; j++) {
    int np = start + j;
    nb[j] = g + 2 * np;
    rb[j] = np - p + 6;
  }
}

__global__ __launch_bounds__(256) void k_na_attn(const u16* __restrict__ qkv2,
                                                 const float* __restrict__ rpb,
                                                 u16* __restrict__ out) {
  int head = blockIdx.y;
  int l = blockIdx.x * blockDim.x + threadIdx.x;
  int h = l >> 8, w = l & 255;
  int nh[7], nw[7], rh[7], rw[7];
  nbr7(h, nh, rh);
  nbr7(w, nw, rw);

  const u16* qp = qkv2 + (size_t)l * 576 + head * 32;
  float q[32];
  {
    const uint4* qu = (const uint4*)qp;
    #pragma unroll
    for (int i = 0; i < 4; i++) {
      uint4 t = qu[i];
      q[i * 8 + 0] = bflo(t.x); q[i * 8 + 1] = bfhi(t.x);
      q[i * 8 + 2] = bflo(t.y); q[i * 8 + 3] = bfhi(t.y);
      q[i * 8 + 4] = bflo(t.z); q[i * 8 + 5] = bfhi(t.z);
      q[i * 8 + 6] = bflo(t.w); q[i * 8 + 7] = bfhi(t.w);
    }
    #pragma unroll
    for (int i = 0; i < 32; i++) q[i] *= 0.17677669529663689f; // 32^-0.5
  }

  float m = -1e30f, s = 0.0f, o[32];
  #pragma unroll
  for (int i = 0; i < 32; i++) o[i] = 0.0f;
  const float* rb = rpb + head * 169;  // (2K-1)^2 = 13*13

  for (int ki = 0; ki < 7; ki++) {
    int rowl = nh[ki] * 256;
    for (int kj = 0; kj < 7; kj++) {
      int kl = rowl + nw[kj];
      const u16* kp = qkv2 + (size_t)kl * 576 + 192 + head * 32;
      if (kj < 6)
        __builtin_prefetch(qkv2 + (size_t)(rowl + nw[kj + 1]) * 576 + 192 + head * 32, 0, 1);
      float dot = 0.0f;
      const uint4* ku = (const uint4*)kp;
      #pragma unroll
      for (int i = 0; i < 4; i++) {
        uint4 t = ku[i];
        dot += q[i * 8 + 0] * bflo(t.x) + q[i * 8 + 1] * bfhi(t.x);
        dot += q[i * 8 + 2] * bflo(t.y) + q[i * 8 + 3] * bfhi(t.y);
        dot += q[i * 8 + 4] * bflo(t.z) + q[i * 8 + 5] * bfhi(t.z);
        dot += q[i * 8 + 6] * bflo(t.w) + q[i * 8 + 7] * bfhi(t.w);
      }
      float logit = dot + rb[rh[ki] * 13 + rw[kj]];
      float mn = fmaxf(m, logit);
      float corr = __expf(m - mn);
      float p = __expf(logit - mn);
      s = s * corr + p;
      const uint4* vu = (const uint4*)(kp + 192);
      #pragma unroll
      for (int i = 0; i < 4; i++) {
        uint4 t = vu[i];
        o[i * 8 + 0] = o[i * 8 + 0] * corr + p * bflo(t.x);
        o[i * 8 + 1] = o[i * 8 + 1] * corr + p * bfhi(t.x);
        o[i * 8 + 2] = o[i * 8 + 2] * corr + p * bflo(t.y);
        o[i * 8 + 3] = o[i * 8 + 3] * corr + p * bfhi(t.y);
        o[i * 8 + 4] = o[i * 8 + 4] * corr + p * bflo(t.z);
        o[i * 8 + 5] = o[i * 8 + 5] * corr + p * bfhi(t.z);
        o[i * 8 + 6] = o[i * 8 + 6] * corr + p * bflo(t.w);
        o[i * 8 + 7] = o[i * 8 + 7] * corr + p * bfhi(t.w);
      }
      m = mn;
    }
  }
  float inv = 1.0f / s;
  u16* op = out + (size_t)l * DIM + head * 32;
  #pragma unroll
  for (int i = 0; i < 32; i++) op[i] = f2bf(o[i] * inv);
}

// -------------------------- CPB MLP: 2->512->6 ----------------------------
__global__ __launch_bounds__(256) void k_cpb(const float* __restrict__ table,
                                             const float* __restrict__ w1a, const float* __restrict__ b1a,
                                             const float* __restrict__ w2a,
                                             const float* __restrict__ w1b, const float* __restrict__ b1b,
                                             const float* __restrict__ w2b,
                                             float* __restrict__ bt) {  // bt[2][225*6]
  __shared__ float acc[HEADS];
  int r = blockIdx.x;          // 0..224
  int tbl = blockIdx.y;        // 0..1
  const float* w1 = tbl ? w1b : w1a;
  const float* b1 = tbl ? b1b : b1a;
  const float* w2 = tbl ? w2b : w2a;
  int tid = threadIdx.x;
  if (tid < HEADS) acc[tid] = 0.0f;
  __syncthreads();
  float t0 = table[2 * r], t1 = table[2 * r + 1];
  for (int hcur = tid; hcur < 512; hcur += 256) {
    float hv = fmaxf(0.0f, t0 * w1[hcur] + t1 * w1[512 + hcur] + b1[hcur]);
    #pragma unroll
    for (int hd = 0; hd < HEADS; hd++) atomicAdd(&acc[hd], hv * w2[hcur * HEADS + hd]);
  }
  __syncthreads();
  if (tid < HEADS) bt[tbl * 225 * HEADS + r * HEADS + tid] = acc[tid];
}

__global__ __launch_bounds__(256) void k_bias_fill(const float* __restrict__ bt,
                                                   const int* __restrict__ idx_a2w,
                                                   const int* __restrict__ idx_w2a,
                                                   float* __restrict__ bias1,   // [6][16][64]
                                                   float* __restrict__ bias2) { // [6][64][16]
  int gid = blockIdx.x * blockDim.x + threadIdx.x;
  if (gid < 6144) {
    int head = gid / 1024, rem = gid % 1024, i = rem / 64, j = rem % 64;
    int idx = idx_a2w[i * 64 + j];
    bias1[gid] = 16.0f / (1.0f + __expf(-bt[idx * HEADS + head]));
  } else if (gid < 12288) {
    int g2 = gid - 6144;
    int head = g2 / 1024, rem = g2 % 1024, i = rem / 16, j = rem % 16;
    int idx = idx_w2a[i * 16 + j];
    bias2[g2] = 16.0f / (1.0f + __expf(-bt[225 * HEADS + idx * HEADS + head]));
  }
}

// ---------------------- stripe (window) attention -------------------------
__device__ __forceinline__ float dot16s(const float* a, const float* b) {
  float s = 0.0f;
  #pragma unroll
  for (int i = 0; i < 16; i++) s += a[i] * b[i];
  return s;
}

__global__ __launch_bounds__(64) void k_stripe(const u16* __restrict__ qkv_bf,
                                               const float* __restrict__ anchor,
                                               const float* __restrict__ bias1,
                                               const float* __restrict__ bias2,
                                               const float* __restrict__ ls1,
                                               const float* __restrict__ ls2,
                                               u16* __restrict__ concat) {
  __shared__ float kn[64][17];
  __shared__ float vs[64][17];
  __shared__ float ann[16][17];
  __shared__ float x1s[16][17];
  int wid = blockIdx.x, head = blockIdx.y, t = threadIdx.x;

  int h = (wid >> 5) * 8 + (t >> 3);
  int w = (wid & 31) * 8 + (t & 7);
  int l = h * 256 + w;
  const u16* base = qkv_bf + (size_t)l * 576 + 288 + head * 16;

  float qn[16];
  {
    float kv[16];
    float nq = 0.0f, nk = 0.0f;
    #pragma unroll
    for (int c = 0; c < 16; c++) {
      qn[c] = bf2f(base[c]);
      kv[c] = bf2f(base[96 + c]);
      vs[t][c] = bf2f(base[192 + c]);
      nq += qn[c] * qn[c];
      nk += kv[c] * kv[c];
    }
    float iq = 1.0f / fmaxf(sqrtf(nq), 1e-12f);
    float ik = 1.0f / fmaxf(sqrtf(nk), 1e-12f);
    #pragma unroll
    for (int c = 0; c < 16; c++) { qn[c] *= iq; kn[t][c] = kv[c] * ik; }
  }
  if (t < 16) {
    int ah = (wid >> 5) * 4 + (t >> 2);
    int aw = (wid & 31) * 4 + (t & 3);
    const float* ap = anchor + (size_t)(ah * 128 + aw) * 96 + head * 16;
    float na = 0.0f, av[16];
    #pragma unroll
    for (int c = 0; c < 16; c++) { av[c] = ap[c]; na += av[c] * av[c]; }
    float ia = 1.0f / fmaxf(sqrtf(na), 1e-12f);
    #pragma unroll
    for (int c = 0; c < 16; c++) ann[t][c] = av[c] * ia;
  }
  __syncthreads();

  float sc1 = __expf(fminf(ls1[head], LOG100));
  if (t < 16) {
    float lg[64], mx = -1e30f;
    const float* b1 = bias1 + (head * 16 + t) * 64;
    for (int j = 0; j < 64; j++) {
      float s = sc1 * dot16s(ann[t], kn[j]) + b1[j];
      lg[j] = s;
      mx = fmaxf(mx, s);
    }
    float sum = 0.0f;
    for (int j = 0; j < 64; j++) { lg[j] = __expf(lg[j] - mx); sum += lg[j]; }
    float inv = 1.0f / sum;
    #pragma unroll
    for (int c = 0; c < 16; c++) {
      float a = 0.0f;
      for (int j = 0; j < 64; j++) a += lg[j] * vs[j][c];
      x1s[t][c] = a * inv;
    }
  }
  __syncthreads();

  float sc2 = __expf(fminf(ls2[head], LOG100));
  float lg[16], mx = -1e30f;
  const float* b2 = bias2 + (head * 64 + t) * 16;
  #pragma unroll
  for (int j = 0; j < 16; j++) {
    float s = sc2 * dot16s(qn, ann[j]) + b2[j];
    lg[j] = s;
    mx = fmaxf(mx, s);
  }
  float sum = 0.0f;
  #pragma unroll
  for (int j = 0; j < 16; j++) { lg[j] = __expf(lg[j] - mx); sum += lg[j]; }
  float inv = 1.0f / sum;
  u16* op = concat + (size_t)l * DIM + 96 + head * 16;
  #pragma unroll
  for (int c = 0; c < 16; c++) {
    float a = 0.0f;
    #pragma unroll
    for (int j = 0; j < 16; j++) a += lg[j] * x1s[j][c];
    op[c] = f2bf(a * inv);
  }
}

// ------------------------------- launcher ---------------------------------
extern "C" void kernel_launch(void* const* d_in, const int* in_sizes, int n_in,
                              void* d_out, int out_size, void* d_ws, size_t ws_size,
                              hipStream_t stream) {
  (void)in_sizes; (void)n_in; (void)out_size; (void)ws_size;
  const float* x       = (const float*)d_in[0];
  const float* table   = (const float*)d_in[3];
  const int*   idx_a2w = (const int*)d_in[4];
  const int*   idx_w2a = (const int*)d_in[5];
  const float* qkv_w   = (const float*)d_in[6];
  const float* qkv_b   = (const float*)d_in[7];
  const float* dw_w    = (const float*)d_in[8];
  const float* dw_b    = (const float*)d_in[9];
  const float* pw_w    = (const float*)d_in[10];
  const float* pw_b    = (const float*)d_in[11];
  const float* naqkv_w = (const float*)d_in[12];
  const float* naqkv_b = (const float*)d_in[13];
  const float* naaft_w = (const float*)d_in[14];
  const float* naaft_b = (const float*)d_in[15];
  const float* na_rpb  = (const float*)d_in[16];
  const float* ls1     = (const float*)d_in[17];
  const float* cpb1_w1 = (const float*)d_in[18];
  const float* cpb1_b1 = (const float*)d_in[19];
  const float* cpb1_w2 = (const float*)d_in[20];
  const float* ls2     = (const float*)d_in[21];
  const float* cpb2_w1 = (const float*)d_in[22];
  const float* cpb2_b1 = (const float*)d_in[23];
  const float* cpb2_w2 = (const float*)d_in[24];
  const float* proj_w  = (const float*)d_in[25];
  const float* proj_b  = (const float*)d_in[26];
  float* out = (float*)d_out;

  char* ws = (char*)d_ws;
  size_t off = 0;
  auto take = [&](size_t bytes) -> char* {
    size_t a = (off + 255) & ~(size_t)255;
    off = a + bytes;
    return ws + a;
  };
  u16* x_bf       = (u16*)take((size_t)LTOK * DIM * 2);
  u16* qkv_bf     = (u16*)take((size_t)LTOK * 576 * 2);
  u16* qkv2_bf    = (u16*)take((size_t)LTOK * 576 * 2);
  u16* naout_bf   = (u16*)take((size_t)LTOK * DIM * 2);
  u16* concat_bf  = (u16*)take((size_t)LTOK * DIM * 2);
  float* anchor_f = (float*)take((size_t)16384 * 96 * 4);
  u16* qkvw_t     = (u16*)take((size_t)576 * 192 * 2);   // [Npad=576][K=192]
  u16* naqkvw_t   = (u16*)take((size_t)576 * 288 * 2);   // [Npad=576][K=288]
  u16* naaftw_t   = (u16*)take((size_t)128 * 192 * 2);   // [Npad=128][K=192]
  u16* projw_t    = (u16*)take((size_t)192 * 192 * 2);   // [Npad=192][K=192]
  float* bt       = (float*)take((size_t)2 * 225 * HEADS * 4);
  float* bias1    = (float*)take((size_t)HEADS * 16 * 64 * 4);
  float* bias2    = (float*)take((size_t)HEADS * 64 * 16 * 4);

  // 1) converts (activations plain; weights transposed+padded to N-major)
  k_convert<<<dim3((LTOK * DIM + 255) / 256), 256, 0, stream>>>(x, x_bf, LTOK * DIM);
  k_convert_T<<<dim3((576 * 192 + 255) / 256), 256, 0, stream>>>(qkv_w, qkvw_t, 192, 576, 576);
  k_convert_T<<<dim3((576 * 288 + 255) / 256), 256, 0, stream>>>(naqkv_w, naqkvw_t, 288, 576, 576);
  k_convert_T<<<dim3((128 * 192 + 255) / 256), 256, 0, stream>>>(naaft_w, naaftw_t, 192, 96, 128);
  k_convert_T<<<dim3((192 * 192 + 255) / 256), 256, 0, stream>>>(proj_w, projw_t, 192, 192, 192);

  // 2) qkv GEMM: [65536,192] x [192,576] -> qkv_bf
  k_gemm_bf16<192><<<dim3(LTOK / BM, 576 / BN), 256, 0, stream>>>(
      x_bf, DIM, qkvw_t, qkv_b, nullptr, qkv_bf, 576, LTOK, 576);

  // 3) anchor conv (dw 3x3 s2 + pw 192->96)
  k_anchor<<<dim3(16384), 192, 0, stream>>>(x, dw_w, dw_b, pw_w, pw_b, anchor_f);

  // 4) na_qkv GEMM: window half of qkv (cols 0..287) x [288,576] -> qkv2_bf
  k_gemm_bf16<288><<<dim3(LTOK / BM, 576 / BN), 256, 0, stream>>>(
      qkv_bf, 576, naqkvw_t, naqkv_b, nullptr, qkv2_bf, 576, LTOK, 576);

  // 5) neighborhood attention -> naout_bf [65536,192]
  k_na_attn<<<dim3(LTOK / 256, HEADS), 256, 0, stream>>>(qkv2_bf, na_rpb, naout_bf);

  // 6) na_after GEMM: [65536,192] x [192,96] -> concat cols 0..95
  k_gemm_bf16<192><<<dim3(LTOK / BM, 128 / BN), 256, 0, stream>>>(
      naout_bf, DIM, naaftw_t, naaft_b, nullptr, concat_bf, DIM, LTOK, 96);

  // 7) CPB MLP + bias tables
  k_cpb<<<dim3(225, 2), 256, 0, stream>>>(table, cpb1_w1, cpb1_b1, cpb1_w2,
                                          cpb2_w1, cpb2_b1, cpb2_w2, bt);
  k_bias_fill<<<dim3(48), 256, 0, stream>>>(bt, idx_a2w, idx_w2a, bias1, bias2);

  // 8) stripe attention -> concat cols 96..191
  k_stripe<<<dim3(1024, HEADS), 64, 0, stream>>>(qkv_bf, anchor_f, bias1, bias2,
                                                 ls1, ls2, concat_bf);

  // 9) proj GEMM: [65536,192] x [192,192] + b -> d_out (f32)
  k_gemm_bf16<192><<<dim3(LTOK / BM, 192 / BN), 256, 0, stream>>>(
      concat_bf, DIM, projw_t, proj_b, out, nullptr, DIM, LTOK, 192);
}
